// FCMoE_cherma_4398046511348
// MI455X (gfx1250) — matrix operand order repaired and checked by reading, not compiled, not run on hardware
//
#include <hip/hip_runtime.h>
#include <hip/hip_bf16.h>
#include <stdint.h>

// ============================================================================
// FCMoE forward for MI455X (gfx1250, wave32, WMMA).
// All dense GEMMs use v_wmma_f32_16x16x32_f16 with f32 accumulate.
// Weights pre-packed to f16 [N][K]; B tiles copied global->LDS with
// global_load_async_to_lds_b128 (ASYNCcnt, zero VGPR transit); A staged
// through registers with cvt_pkrtz f32->f16. Block tile 128x128, wave tile
// 32x64 (8 WMMAs/wave/k-step), double-buffered LDS, 1 barrier per k-step.
// All GEMM shapes satisfy M%128==0, N%128==0, K%32==0 -> no edge checks.
// ============================================================================

typedef __attribute__((ext_vector_type(16))) _Float16 v16h;
typedef __attribute__((ext_vector_type(8)))  float    v8f;

#define BCONST 256
#define DMODEL 256
#define NHEAD  8

__device__ __forceinline__ unsigned short f2h(float f) {
    union { _Float16 h; unsigned short u; } c;
    c.h = (_Float16)f;
    return c.u;
}

// pack two f32 -> one u32 of two f16 (v_cvt_pk_rtz_f16_f32)
__device__ __forceinline__ uint32_t pk2h(float x, float y) {
    union { __fp16 h2 __attribute__((ext_vector_type(2))); uint32_t u; } c;
    c.h2 = __builtin_amdgcn_cvt_pkrtz(x, y);
    return c.u;
}

__device__ __forceinline__ float wave_sum(float v) {
#pragma unroll
    for (int o = 16; o > 0; o >>= 1) v += __shfl_xor(v, o, 32);
    return v;
}

// ---------------------------------------------------------------------------
// weight pack: W[K][N] f32 -> Wp[N][K] f16
// ---------------------------------------------------------------------------
__global__ void k_pack_w(const float* __restrict__ W, unsigned short* __restrict__ Wp,
                         int K, int N)
{
    long n = (long)K * N;
    long stride = (long)gridDim.x * blockDim.x;
    for (long i = (long)blockIdx.x*blockDim.x + threadIdx.x; i < n; i += stride) {
        int nn = (int)(i / K);
        int kk = (int)(i - (long)nn*K);
        Wp[i] = f2h(W[(long)kk*N + nn]);
    }
}

// conv weight rearrange+pack: w[O][IC][3] f32 -> wtp[O][kk*IC+ic] f16  ([N][K])
__global__ void k_conv_wt(const float* __restrict__ w, unsigned short* __restrict__ wtp, int IC)
{
    long n = (long)DMODEL * IC * 3;
    long stride = (long)gridDim.x * blockDim.x;
    for (long i = (long)blockIdx.x*blockDim.x + threadIdx.x; i < n; i += stride) {
        int o   = (int)(i / (3*IC));
        int rem = (int)(i - (long)o*3*IC);
        int kk  = rem / IC, ic = rem - kk*IC;
        wtp[i] = f2h(w[((long)o*IC + ic)*3 + kk]);
    }
}

// ---------------------------------------------------------------------------
// WMMA GEMM:  C[M,N] = act(A[M,K] @ W[K,N] + bias) (+ resid), W given as f16 [N][K]
// 256 threads = 8 waves in 4(M) x 2(N); wave tile 32x64 -> 8 WMMAs per k-step.
// Double-buffered LDS; B via async global->LDS DMA; software-pipelined.
// IM2COL=1: A is implicit im2col of x[B][Lin][IC], conv k=3 pad=1, K=3*IC
//           (conv GEMM has no bias/resid). HAS_BIAS == !IM2COL.
// ---------------------------------------------------------------------------
#define GBM 128
#define GBN 128
#define GBK 32
#define LDSS 40   // padded LDS row stride in halves: rows stay 16B aligned

union FragU { v16h v; uint32_t u[8]; };

template<int IM2COL, int RESID>
__global__ __launch_bounds__(256)
void k_gemm(const float* __restrict__ A, const unsigned short* __restrict__ Wp,
            const float* __restrict__ bias, const float* __restrict__ resid,
            float* __restrict__ C,
            int M, int N, int K, int lda, int ldc, int act,
            int L, int Lin, int IC)
{
    __shared__ unsigned short As[2][GBM][LDSS];
    __shared__ unsigned short Bs[2][GBN][LDSS];

    const int tid  = threadIdx.x;
    const int m0   = blockIdx.x * GBM;
    const int n0   = blockIdx.y * GBN;
    const int w    = tid >> 5;
    const int lane = tid & 31;
    const int wm = w >> 1, wn = w & 1;       // 4 x 2 wave grid
    const int lr = lane & 15, lh = lane >> 4;

    // staging coordinates: each thread covers 16 consecutive K-elements of one row
    const int sr   = tid >> 1;               // row 0..127 (A: m, B: n)
    const int sc16 = (tid & 1) * 16;         // col base {0,16}
    const unsigned short* wrow = Wp + (long)(n0 + sr)*K + sc16;

    // fragment K offsets
    int kofsA[8];                            // ISA 16-bit A-matrix 16x32 layout
#pragma unroll
    for (int r = 0; r < 8; ++r)
        kofsA[r] = (r < 4) ? (2*r + 8*lh) : (16 + 2*(r-4) + 8*lh);
    int kofsB[8];                            // same k order as A (16-bit 16x32 layout)
#pragma unroll
    for (int r = 0; r < 8; ++r)
        kofsB[r] = (r < 4) ? (2*r + 8*lh) : (16 + 2*(r-4) + 8*lh);

    float4 a4[4];

    auto g_load = [&](int k0, int bufi) {
        // --- B tile: async DMA global f16 -> LDS (no VGPR transit) ---
        const unsigned short* wsrc = wrow + k0;
        uint32_t ldsb = (uint32_t)(uintptr_t)&Bs[bufi][sr][sc16];
        asm volatile("global_load_async_to_lds_b128 %0, %1, off"
                     :: "v"(ldsb), "v"(wsrc) : "memory");
        asm volatile("global_load_async_to_lds_b128 %0, %1, off offset:16"
                     :: "v"(ldsb), "v"(wsrc) : "memory");
        // --- A tile: f32 loads into registers (converted at store time) ---
        if (IM2COL) {
            int gm = m0 + sr;
            int b  = gm / L, l = gm - b*L;
            int gk = k0 + sc16;
            int kk = gk / IC, ic = gk - kk*IC;
            int ls = l + kk - 1;
            bool valid = (ls >= 0) && (ls < L);
            int lsc = valid ? ls : 0;
            const float4* ap = (const float4*)(A + ((long)b*Lin + lsc)*IC + ic);
            float4 t0 = ap[0], t1 = ap[1], t2 = ap[2], t3 = ap[3];
            if (!valid) {
                t0 = make_float4(0.f,0.f,0.f,0.f); t1 = t0; t2 = t0; t3 = t0;
            }
            a4[0] = t0; a4[1] = t1; a4[2] = t2; a4[3] = t3;
        } else {
            const float* arow = A + (long)(m0 + sr)*lda + k0 + sc16;
            const float4* ap = (const float4*)arow;
            a4[0] = ap[0]; a4[1] = ap[1]; a4[2] = ap[2]; a4[3] = ap[3];
            if (k0 + 2*GBK <= K - GBK)
                __builtin_prefetch(arow + 2*GBK, 0, 1);
        }
    };
    auto s_store = [&](int bufi) {
        uint4 p0, p1;
        p0.x = pk2h(a4[0].x, a4[0].y);  p0.y = pk2h(a4[0].z, a4[0].w);
        p0.z = pk2h(a4[1].x, a4[1].y);  p0.w = pk2h(a4[1].z, a4[1].w);
        p1.x = pk2h(a4[2].x, a4[2].y);  p1.y = pk2h(a4[2].z, a4[2].w);
        p1.z = pk2h(a4[3].x, a4[3].y);  p1.w = pk2h(a4[3].z, a4[3].w);
        *(uint4*)&As[bufi][sr][sc16]     = p0;
        *(uint4*)&As[bufi][sr][sc16 + 8] = p1;
    };

    v8f acc[2][4] = {};
    const int nk = K / GBK;

    g_load(0, 0);
    s_store(0);
    asm volatile("s_wait_asynccnt 0" ::: "memory");
    __syncthreads();

    const int ar0 = wm*32 + lr, ar1 = wm*32 + 16 + lr;
    int bcn[4];
#pragma unroll
    for (int ni = 0; ni < 4; ++ni) bcn[ni] = wn*64 + ni*16 + lr;

    for (int kt = 0; kt < nk; ++kt) {
        const bool more = (kt + 1 < nk);
        const int cur = kt & 1;
        // issue next tile: B async into the other LDS buffer, A into registers
        if (more) g_load((kt + 1) * GBK, 1 - cur);

        FragU af0, af1, bf[4];
#pragma unroll
        for (int r = 0; r < 8; ++r) {
            af0.u[r] = *(const uint32_t*)&As[cur][ar0][kofsA[r]];
            af1.u[r] = *(const uint32_t*)&As[cur][ar1][kofsA[r]];
        }
#pragma unroll
        for (int ni = 0; ni < 4; ++ni)
#pragma unroll
            for (int r = 0; r < 8; ++r)
                bf[ni].u[r] = *(const uint32_t*)&Bs[cur][bcn[ni]][kofsB[r]];

#pragma unroll
        for (int ni = 0; ni < 4; ++ni) {
            acc[0][ni] = __builtin_amdgcn_wmma_f32_16x16x32_f16(false, af0.v, false, bf[ni].v, (short)0, acc[0][ni], false, false);
            acc[1][ni] = __builtin_amdgcn_wmma_f32_16x16x32_f16(false, af1.v, false, bf[ni].v, (short)0, acc[1][ni], false, false);
        }
        if (more) s_store(1 - cur);          // write other buffer; readers use cur
        asm volatile("s_wait_asynccnt 0" ::: "memory");
        __syncthreads();
    }

    // epilogue: D layout -> row_local = 8*lh + r, col_local = lr
    float bv[4];
#pragma unroll
    for (int ni = 0; ni < 4; ++ni)
        bv[ni] = (IM2COL == 0) ? bias[n0 + wn*64 + ni*16 + lr] : 0.f;

#pragma unroll
    for (int mi = 0; mi < 2; ++mi) {
#pragma unroll
        for (int ni = 0; ni < 4; ++ni) {
            const int col = n0 + wn*64 + ni*16 + lr;
#pragma unroll
            for (int r = 0; r < 8; ++r) {
                int row = m0 + wm*32 + mi*16 + 8*lh + r;
                float v = acc[mi][ni][r] + bv[ni];
                if (act) v = fmaxf(v, 0.f);
                if (RESID) v += resid[(long)row*ldc + col];
                C[(long)row*ldc + col] = v;
            }
        }
    }
}

// ---------------------------------------------------------------------------
// h = h*sqrt(D) + sinpos(l, d)
// ---------------------------------------------------------------------------
__global__ void k_addpos(float* __restrict__ H, long T, int L)
{
    long n = T * DMODEL;
    long stride = (long)gridDim.x * blockDim.x;
    for (long i = (long)blockIdx.x*blockDim.x + threadIdx.x; i < n; i += stride) {
        int d = (int)(i & 255);
        long t = i >> 8;
        int l = (int)(t % L);
        int j = d & 127;
        float freq = __expf((float)j * (-9.210340371976184f / 127.f));
        float ang = (float)l * freq;
        float pe = (d < 128) ? __sinf(ang) : __cosf(ang);
        H[i] = H[i] * 16.0f + pe;
    }
}

// ---------------------------------------------------------------------------
// LayerNorm rows of 256; one wave per row, float4 I/O (8 contiguous / lane)
// ---------------------------------------------------------------------------
__global__ __launch_bounds__(256)
void k_layernorm(const float* __restrict__ X, const float* __restrict__ g,
                 const float* __restrict__ be, float* __restrict__ Y, long T)
{
    long row = (long)blockIdx.x*8 + (threadIdx.x >> 5);
    if (row >= T) return;
    int lane = threadIdx.x & 31;
    const float4* x4 = (const float4*)(X + row*DMODEL + lane*8);
    float4 v0 = x4[0], v1 = x4[1];
    float s = v0.x+v0.y+v0.z+v0.w + v1.x+v1.y+v1.z+v1.w;
    s = wave_sum(s);
    float mean = s * (1.f/256.f);
    float q = (v0.x-mean)*(v0.x-mean) + (v0.y-mean)*(v0.y-mean)
            + (v0.z-mean)*(v0.z-mean) + (v0.w-mean)*(v0.w-mean)
            + (v1.x-mean)*(v1.x-mean) + (v1.y-mean)*(v1.y-mean)
            + (v1.z-mean)*(v1.z-mean) + (v1.w-mean)*(v1.w-mean);
    q = wave_sum(q);
    float rstd = rsqrtf(q*(1.f/256.f) + 1e-5f);
    const float4* g4 = (const float4*)(g + lane*8);
    const float4* b4 = (const float4*)(be + lane*8);
    float4 G0 = g4[0], G1 = g4[1], B0 = b4[0], B1 = b4[1];
    float4 o0, o1;
    o0.x = (v0.x-mean)*rstd*G0.x + B0.x;  o0.y = (v0.y-mean)*rstd*G0.y + B0.y;
    o0.z = (v0.z-mean)*rstd*G0.z + B0.z;  o0.w = (v0.w-mean)*rstd*G0.w + B0.w;
    o1.x = (v1.x-mean)*rstd*G1.x + B1.x;  o1.y = (v1.y-mean)*rstd*G1.y + B1.y;
    o1.z = (v1.z-mean)*rstd*G1.z + B1.z;  o1.w = (v1.w-mean)*rstd*G1.w + B1.w;
    float4* y4 = (float4*)(Y + row*DMODEL + lane*8);
    y4[0] = o0; y4[1] = o1;
}

// ---------------------------------------------------------------------------
// MHA core: per (b,h), scores->softmax->AV. hd=32. Q/K/V: [B*S][256], head at +h*32
// ---------------------------------------------------------------------------
__global__ __launch_bounds__(128)
void k_attention(const float* __restrict__ Q, const float* __restrict__ Kb,
                 const float* __restrict__ Vb, float* __restrict__ O, int S)
{
    __shared__ float sc[256];
    __shared__ float qrow[32];
    __shared__ float red[128];
    __shared__ float part[4][32];
    int b = blockIdx.x >> 3, h = blockIdx.x & 7;
    const float scale = 0.17677669529663687f;   // 1/sqrt(32)
    long base = ((long)b * S) * DMODEL + h*32;
    int tid = threadIdx.x;

    for (int qi = 0; qi < S; ++qi) {
        if (tid < 32) qrow[tid] = Q[base + (long)qi*DMODEL + tid];
        __syncthreads();
        for (int k = tid; k < S; k += 128) {
            const float4* kp = (const float4*)(Kb + base + (long)k*DMODEL);
            float d = 0.f;
#pragma unroll
            for (int j = 0; j < 8; ++j) {
                float4 kv = kp[j];
                d += qrow[4*j+0]*kv.x + qrow[4*j+1]*kv.y
                   + qrow[4*j+2]*kv.z + qrow[4*j+3]*kv.w;
            }
            sc[k] = d*scale;
        }
        __syncthreads();
        float m = -1e30f;
        for (int k = tid; k < S; k += 128) m = fmaxf(m, sc[k]);
        red[tid] = m; __syncthreads();
        for (int s2 = 64; s2 > 0; s2 >>= 1) {
            if (tid < s2) red[tid] = fmaxf(red[tid], red[tid+s2]);
            __syncthreads();
        }
        m = red[0]; __syncthreads();
        float sum = 0.f;
        for (int k = tid; k < S; k += 128) { float e = __expf(sc[k]-m); sc[k] = e; sum += e; }
        red[tid] = sum; __syncthreads();
        for (int s2 = 64; s2 > 0; s2 >>= 1) {
            if (tid < s2) red[tid] += red[tid+s2];
            __syncthreads();
        }
        float inv = 1.f/red[0]; __syncthreads();
        int d = tid & 31, p = tid >> 5;
        float o = 0.f;
        for (int k = p; k < S; k += 4) o += sc[k]*Vb[base + (long)k*DMODEL + d];
        part[p][d] = o; __syncthreads();
        if (tid < 32) {
            float r = (part[0][tid]+part[1][tid]+part[2][tid]+part[3][tid])*inv;
            O[base + (long)qi*DMODEL + tid] = r;
        }
        __syncthreads();
    }
}

// ---------------------------------------------------------------------------
// coarse/fine max pooling over seq. X:[B*L][256]. fine:[B][256][gran]
// ---------------------------------------------------------------------------
__global__ void k_pool_max(const float* __restrict__ X, float* __restrict__ coarse,
                           float* __restrict__ fine, int L, int gran)
{
    int i = blockIdx.x*blockDim.x + threadIdx.x;
    if (i >= BCONST*DMODEL) return;
    int b = i >> 8, d = i & 255;
    int seg = L / gran;
    float cm = -1e30f;
    for (int g = 0; g < gran; ++g) {
        float fm = -1e30f;
        for (int s = 0; s < seg; ++s) {
            float v = X[((long)b*L + g*seg + s)*DMODEL + d];
            fm = fmaxf(fm, v);
        }
        fine[((long)b*DMODEL + d)*gran + g] = fm;
        cm = fmaxf(cm, fm);
    }
    coarse[(long)b*DMODEL + d] = cm;
}

// ---------------------------------------------------------------------------
// MoE gating: softmax(x@Wg) -> top2 -> normalized gates [B][8]
// ---------------------------------------------------------------------------
__global__ __launch_bounds__(256)
void k_moe_gate(const float* __restrict__ X, const float* __restrict__ Wg,
                float* __restrict__ gates, int in_dim)
{
    __shared__ float logit[8];
    int b = blockIdx.x;
    int e = threadIdx.x >> 5, lane = threadIdx.x & 31;
    const float* x = X + (long)b*in_dim;
    float s = 0.f;
    for (int i = lane; i < in_dim; i += 32) s += x[i]*Wg[(long)i*8 + e];
    s = wave_sum(s);
    if (lane == 0) logit[e] = s;
    __syncthreads();
    if (threadIdx.x == 0) {
        float mx = -1e30f;
        for (int i = 0; i < 8; ++i) mx = fmaxf(mx, logit[i]);
        float p[8]; float den = 0.f;
        for (int i = 0; i < 8; ++i) { p[i] = __expf(logit[i]-mx); den += p[i]; }
        for (int i = 0; i < 8; ++i) p[i] /= den;
        int i0 = -1, i1 = -1; float v0 = -1.f, v1 = -1.f;
        for (int i = 0; i < 8; ++i) {
            if (p[i] > v0)      { v1 = v0; i1 = i0; v0 = p[i]; i0 = i; }
            else if (p[i] > v1) { v1 = p[i]; i1 = i; }
        }
        float gs = v0 + v1 + 1e-6f;
        for (int i = 0; i < 8; ++i) gates[b*8+i] = 0.f;
        gates[b*8+i0] = v0/gs;
        gates[b*8+i1] = v1/gs;
    }
}

__global__ void k_moe_combine(const float* __restrict__ Ye, const float* __restrict__ gates,
                              float* __restrict__ Y)
{
    int i = blockIdx.x*blockDim.x + threadIdx.x;
    if (i >= BCONST*DMODEL) return;
    int b = i >> 8;
    float s = 0.f;
#pragma unroll
    for (int e = 0; e < 8; ++e) {
        float g = gates[b*8+e];
        if (g != 0.f) s += g * Ye[(long)e*BCONST*DMODEL + i];
    }
    Y[i] = s;
}

__device__ __forceinline__ float cv2_8(const float* x) {
    float s = 0.f; for (int i = 0; i < 8; ++i) s += x[i];
    float m = s / 8.f;
    float q = 0.f; for (int i = 0; i < 8; ++i) { float d = x[i]-m; q += d*d; }
    float var = q / 7.f;
    return var / (m*m + 1e-10f);
}

__global__ __launch_bounds__(256)
void k_moe_loss(const float* __restrict__ gates, float* __restrict__ loss_out)
{
    __shared__ float imp[8], ld[8];
    int e = threadIdx.x >> 5, lane = threadIdx.x & 31;
    float si = 0.f, sl = 0.f;
    for (int b = lane; b < BCONST; b += 32) {
        float g = gates[b*8+e];
        si += g;
        sl += (g > 0.f) ? 1.f : 0.f;
    }
    si = wave_sum(si); sl = wave_sum(sl);
    if (lane == 0) { imp[e] = si; ld[e] = sl; }
    __syncthreads();
    if (threadIdx.x == 0)
        atomicAdd(loss_out, 0.01f*(cv2_8(imp) + cv2_8(ld)));
}

__global__ void k_set_zero(float* p) { if (threadIdx.x == 0 && blockIdx.x == 0) p[0] = 0.f; }

// ---------------------------------------------------------------------------
// fusion stage helpers
// ---------------------------------------------------------------------------
__global__ void k_stack6(const float* ct, const float* fts, const float* ca,
                         const float* fas, const float* cvv, const float* fvs,
                         float* __restrict__ allf)
{
    int i = blockIdx.x*blockDim.x + threadIdx.x;
    if (i >= BCONST*6*DMODEL) return;
    int d = i & 255; int r = i >> 8; int j = r % 6; int b = r / 6;
    const float* src;
    switch (j) {
        case 0: src = ct;  break; case 1: src = fts; break;
        case 2: src = ca;  break; case 3: src = fas; break;
        case 4: src = cvv; break; default: src = fvs; break;
    }
    allf[i] = src[b*DMODEL + d];
}

__global__ void k_gather3(const float* __restrict__ h, float* __restrict__ cat,
                          int i0, int i1, int i2)
{
    int i = blockIdx.x*blockDim.x + threadIdx.x;
    if (i >= BCONST*768) return;
    int d = i % 768; int b = i / 768;
    int j = d >> 8, dd = d & 255;
    int idx = (j == 0) ? i0 : ((j == 1) ? i1 : i2);
    cat[i] = h[((long)b*6 + idx)*DMODEL + dd];
}

__global__ __launch_bounds__(96)
void k_attnscore(const float* __restrict__ m, const float* __restrict__ Wa,
                 const float* __restrict__ ba, float* __restrict__ aw)
{
    __shared__ float av[3];
    int b = blockIdx.x;
    int j = threadIdx.x >> 5, lane = threadIdx.x & 31;
    float s = 0.f;
    for (int d = lane; d < DMODEL; d += 32) s += m[b*DMODEL+d]*Wa[d*3+j];
    s = wave_sum(s);
    if (lane == 0) av[j] = s + ba[j];
    __syncthreads();
    if (threadIdx.x == 0) {
        const float sc = 0.0625f;  // D^-0.5
        float a0 = av[0]*sc, a1 = av[1]*sc, a2 = av[2]*sc;
        float mx = fmaxf(a0, fmaxf(a1, a2));
        float e0 = __expf(a0-mx), e1 = __expf(a1-mx), e2 = __expf(a2-mx);
        float den = e0+e1+e2;
        aw[b*3+0] = e0/den; aw[b*3+1] = e1/den; aw[b*3+2] = e2/den;
    }
}

__global__ void k_fuse3(const float* __restrict__ f1, const float* __restrict__ f2,
                        const float* __restrict__ f3, int rs,
                        const float* __restrict__ aw, float* __restrict__ out)
{
    int i = blockIdx.x*blockDim.x + threadIdx.x;
    if (i >= BCONST*DMODEL) return;
    int b = i >> 8, d = i & 255;
    out[i] = f1[(long)b*rs+d]*aw[b*3] + f2[(long)b*rs+d]*aw[b*3+1] + f3[(long)b*rs+d]*aw[b*3+2];
}

__global__ void k_concat2(const float* __restrict__ a, const float* __restrict__ b,
                          float* __restrict__ out)
{
    int i = blockIdx.x*blockDim.x + threadIdx.x;
    if (i >= BCONST*512) return;
    int bi = i >> 9, d = i & 511;
    out[i] = (d < 256) ? a[bi*256+d] : b[bi*256+d-256];
}

__global__ __launch_bounds__(224)
void k_logits(const float* __restrict__ fa, const float* __restrict__ fb,
              const float* __restrict__ Wc, const float* __restrict__ bc,
              float* __restrict__ out)
{
    int b = blockIdx.x;
    int c = threadIdx.x >> 5, lane = threadIdx.x & 31;
    float s = 0.f;
    for (int d = lane; d < 256; d += 32)
        s += fa[b*256+d]*Wc[(long)d*7+c] + fb[b*256+d]*Wc[(long)(256+d)*7+c];
    s = wave_sum(s);
    if (lane == 0) out[b*7+c] = s + bc[c];
}

// ============================================================================
// Host orchestration
// ============================================================================
struct LayerP { const float *Wq,*bq,*Wk,*bk,*Wv,*bv,*Wo,*bo,*ln1g,*ln1b,*W1,*b1,*W2,*b2,*ln2g,*ln2b; };
struct EncP   { LayerP layers[4]; const float *lng,*lnb; int nlayers; };
struct MoeP   { const float *wg,*W1,*b1,*W2,*b2; };
struct AttnP  { const float *Wm,*bm,*Wa,*ba; };

static inline int cdiv(long a, long b) { return (int)((a + b - 1) / b); }

extern "C" void kernel_launch(void* const* d_in, const int* in_sizes, int n_in,
                              void* d_out, int out_size, void* d_ws, size_t ws_size,
                              hipStream_t stream)
{
    (void)in_sizes; (void)n_in; (void)out_size; (void)ws_size;

    // ------- inputs (setup_inputs insertion order) -------
    const float* x_t = (const float*)d_in[0];   // [256][96][1024] (use first 80 rows)
    const float* x_a = (const float*)d_in[1];   // [256][200][1024]
    const float* x_v = (const float*)d_in[2];   // [256][32][2048]
    // d_in[3..6] = labels (unused in forward)

    // ------- params: depth-first, dict-insertion-order leaf flattening -------
    int pi = 7;
    auto nxt = [&]() { return (const float*)d_in[pi++]; };
    auto read_layer = [&](LayerP& L) {
        L.Wq=nxt(); L.bq=nxt(); L.Wk=nxt(); L.bk=nxt(); L.Wv=nxt(); L.bv=nxt();
        L.Wo=nxt(); L.bo=nxt(); L.ln1g=nxt(); L.ln1b=nxt();
        L.W1=nxt(); L.b1=nxt(); L.W2=nxt(); L.b2=nxt(); L.ln2g=nxt(); L.ln2b=nxt();
    };
    auto read_enc = [&](EncP& E, int nl, bool final_ln) {
        E.nlayers = nl;
        for (int l = 0; l < nl; ++l) read_layer(E.layers[l]);
        if (final_ln) { E.lng = nxt(); E.lnb = nxt(); } else { E.lng = nullptr; E.lnb = nullptr; }
    };
    const float* conv_t = nxt();
    const float* conv_a = nxt();
    const float* conv_v = nxt();
    EncP enc_t, enc_a, enc_v, fuse_enc;
    read_enc(enc_t, 4, true);
    read_enc(enc_a, 4, true);
    read_enc(enc_v, 4, true);
    MoeP moe_t, moe_a, moe_v;
    auto read_moe = [&](MoeP& M) { M.wg=nxt(); M.W1=nxt(); M.b1=nxt(); M.W2=nxt(); M.b2=nxt(); };
    read_moe(moe_t); read_moe(moe_a); read_moe(moe_v);
    read_enc(fuse_enc, 2, false);
    AttnP attn_cs, attn_fs;
    auto read_attn = [&](AttnP& A) { A.Wm=nxt(); A.bm=nxt(); A.Wa=nxt(); A.ba=nxt(); };
    read_attn(attn_cs); read_attn(attn_fs);
    const float* fc_ts_W = nxt(); const float* fc_ts_b = nxt();
    const float* fc_as_W = nxt(); const float* fc_as_b = nxt();
    const float* fc_vs_W = nxt(); const float* fc_vs_b = nxt();
    const float* fc_all_W = nxt(); const float* fc_all_b = nxt();
    const float* cls_W = nxt(); const float* cls_b = nxt();

    // ------- workspace carving -------
    float* ws = (float*)d_ws;
    size_t off = 0;
    auto alloc = [&](size_t n) { float* p = ws + off; off += n; return p; };
    const size_t MAXTOK = (size_t)BCONST * 200;
    unsigned short* wtp   = (unsigned short*)alloc((size_t)3*2048*256/2 + 16);  // conv f16 pack
    unsigned short* wpack = (unsigned short*)alloc((size_t)2048*256/2 + 16);    // gemm f16 pack
    float* hA    = alloc(MAXTOK*256);
    float* hB    = alloc(MAXTOK*256);
    float* lnbuf = alloc(MAXTOK*256);
    float* qb    = alloc(MAXTOK*256);
    float* kbuf  = alloc(MAXTOK*256);
    float* vb    = alloc(MAXTOK*256);
    float* ab    = alloc(MAXTOK*256);
    float* midb  = alloc(MAXTOK*1024);
    float* ct    = alloc(65536);
    float* ca    = alloc(65536);
    float* cvb   = alloc(65536);
    float* ft    = alloc((size_t)65536*8);
    float* fa    = alloc((size_t)65536*8);
    float* fv    = alloc((size_t)65536*4);
    float* gatesb= alloc(256*8);
    float* hexp  = alloc((size_t)8*65536);
    float* yexp  = alloc((size_t)8*65536);
    float* fts   = alloc(65536);
    float* fas   = alloc(65536);
    float* fvs   = alloc(65536);
    float* allf  = alloc((size_t)6*65536);
    float* catb  = alloc((size_t)256*768);
    float* mbuf  = alloc(65536);
    float* aw_cs = alloc(1024);
    float* aw_fs = alloc(1024);
    float* a_cs  = alloc(65536);
    float* a_fs  = alloc(65536);
    float* cat2  = alloc((size_t)256*512);
    float* fea_cfs = alloc(65536);
    float* f_tb  = alloc(65536);
    float* f_ab  = alloc(65536);
    float* f_vb  = alloc(65536);
    float* feas  = alloc(65536);

    float* out_loss   = (float*)d_out;
    float* out_logits = (float*)d_out + 1;

    // ------- gemm launcher: pack weight to f16 [N][K], then WMMA gemm -------
    auto gemm = [&](const float* A, const float* W, const float* bias, const float* resid,
                    float* C, long M, int N, int K, int lda, int ldc, int act) {
        long nw = (long)K*N;
        k_pack_w<<<cdiv(nw, 256), 256, 0, stream>>>(W, wpack, K, N);
        dim3 g(cdiv(M, GBM), cdiv(N, GBN));
        if (resid)
            k_gemm<0,1><<<g, 256, 0, stream>>>(A, wpack, bias, resid, C, (int)M, N, K, lda, ldc, act, 0, 0, 0);
        else
            k_gemm<0,0><<<g, 256, 0, stream>>>(A, wpack, bias, resid, C, (int)M, N, K, lda, ldc, act, 0, 0, 0);
    };

    // ------- branch: conv -> +pos -> 4x pre-LN encoder -> final LN -> pool -------
    auto run_branch = [&](const float* x, int Lin, int L, int IC, const float* convw,
                          const EncP& enc, int gran, float* coarse, float* fine) {
        long T = (long)BCONST * L;
        int K3 = 3*IC;
        k_conv_wt<<<4096, 256, 0, stream>>>(convw, wtp, IC);
        {
            dim3 g(cdiv(T, GBM), cdiv(DMODEL, GBN));
            k_gemm<1,0><<<g, 256, 0, stream>>>(x, wtp, nullptr, nullptr, hA,
                                               (int)T, DMODEL, K3, 0, DMODEL, 0,
                                               L, Lin, IC);
        }
        k_addpos<<<cdiv(T*DMODEL, 256), 256, 0, stream>>>(hA, T, L);
        float* h = hA; float* hn = hB;
        for (int li = 0; li < 4; ++li) {
            const LayerP& lp = enc.layers[li];
            k_layernorm<<<cdiv(T, 8), 256, 0, stream>>>(h, lp.ln1g, lp.ln1b, lnbuf, T);
            gemm(lnbuf, lp.Wq, lp.bq, nullptr, qb,   T, 256, 256, 256, 256, 0);
            gemm(lnbuf, lp.Wk, lp.bk, nullptr, kbuf, T, 256, 256, 256, 256, 0);
            gemm(lnbuf, lp.Wv, lp.bv, nullptr, vb,   T, 256, 256, 256, 256, 0);
            k_attention<<<BCONST*NHEAD, 128, 0, stream>>>(qb, kbuf, vb, ab, L);
            gemm(ab, lp.Wo, lp.bo, h, hn, T, 256, 256, 256, 256, 0);
            { float* t = h; h = hn; hn = t; }
            k_layernorm<<<cdiv(T, 8), 256, 0, stream>>>(h, lp.ln2g, lp.ln2b, lnbuf, T);
            gemm(lnbuf, lp.W1, lp.b1, nullptr, midb, T, 1024, 256, 256, 1024, 1);
            gemm(midb, lp.W2, lp.b2, h, hn, T, 256, 1024, 1024, 256, 0);
            { float* t = h; h = hn; hn = t; }
        }
        k_layernorm<<<cdiv(T, 8), 256, 0, stream>>>(h, enc.lng, enc.lnb, lnbuf, T);
        k_pool_max<<<cdiv(BCONST*DMODEL, 256), 256, 0, stream>>>(lnbuf, coarse, fine, L, gran);
    };

    // ------- MoE -------
    auto run_moe = [&](const float* x, int in_dim, const MoeP& mp, float* y) {
        k_moe_gate<<<BCONST, 256, 0, stream>>>(x, mp.wg, gatesb, in_dim);
        for (int e = 0; e < 8; ++e) {
            gemm(x, mp.W1 + (long)e*in_dim*256, mp.b1 + e*256, nullptr,
                 hexp + (long)e*65536, BCONST, 256, in_dim, in_dim, 256, 1);
            gemm(hexp + (long)e*65536, mp.W2 + (long)e*65536, mp.b2 + e*256, nullptr,
                 yexp + (long)e*65536, BCONST, 256, 256, 256, 256, 0);
        }
        k_moe_combine<<<cdiv(BCONST*DMODEL, 256), 256, 0, stream>>>(yexp, gatesb, y);
        k_moe_loss<<<1, 256, 0, stream>>>(gatesb, out_loss);
    };

    // ================= forward =================
    k_set_zero<<<1, 1, 0, stream>>>(out_loss);

    run_branch(x_a, 200, 200, 1024, conv_a, enc_a, 8, ca,  fa);
    run_branch(x_v,  32,  32, 2048, conv_v, enc_v, 4, cvb, fv);
    run_branch(x_t,  96,  80, 1024, conv_t, enc_t, 8, ct,  ft);

    run_moe(ft, 2048, moe_t, fts);
    run_moe(fa, 2048, moe_a, fas);
    run_moe(fv, 1024, moe_v, fvs);

    // fusion encoder (post-LN, 2 layers, S=6, FFN=2048)
    k_stack6<<<cdiv(BCONST*6*DMODEL, 256), 256, 0, stream>>>(ct, fts, ca, fas, cvb, fvs, allf);
    {
        const long T = (long)BCONST * 6;
        float* h = allf;
        for (int li = 0; li < 2; ++li) {
            const LayerP& lp = fuse_enc.layers[li];
            gemm(h, lp.Wq, lp.bq, nullptr, qb,   T, 256, 256, 256, 256, 0);
            gemm(h, lp.Wk, lp.bk, nullptr, kbuf, T, 256, 256, 256, 256, 0);
            gemm(h, lp.Wv, lp.bv, nullptr, vb,   T, 256, 256, 256, 256, 0);
            k_attention<<<BCONST*NHEAD, 128, 0, stream>>>(qb, kbuf, vb, ab, 6);
            gemm(ab, lp.Wo, lp.bo, h, hB, T, 256, 256, 256, 256, 0);
            k_layernorm<<<cdiv(T, 8), 256, 0, stream>>>(hB, lp.ln1g, lp.ln1b, hA, T);
            h = hA;
            gemm(h, lp.W1, lp.b1, nullptr, midb, T, 2048, 256, 256, 2048, 1);
            gemm(midb, lp.W2, lp.b2, h, hB, T, 256, 2048, 2048, 256, 0);
            k_layernorm<<<cdiv(T, 8), 256, 0, stream>>>(hB, lp.ln2g, lp.ln2b, allf, T);
            h = allf;
        }
    }

    // attention fusion (coarse: h0,h2,h4 ; fine: h1,h3,h5)
    k_gather3<<<cdiv(BCONST*768, 256), 256, 0, stream>>>(allf, catb, 0, 2, 4);
    gemm(catb, attn_cs.Wm, attn_cs.bm, nullptr, mbuf, BCONST, 256, 768, 768, 256, 1);
    k_attnscore<<<BCONST, 96, 0, stream>>>(mbuf, attn_cs.Wa, attn_cs.ba, aw_cs);
    k_fuse3<<<cdiv(BCONST*DMODEL, 256), 256, 0, stream>>>(allf + 0*256, allf + 2*256, allf + 4*256,
                                                          1536, aw_cs, a_cs);
    k_gather3<<<cdiv(BCONST*768, 256), 256, 0, stream>>>(allf, catb, 1, 3, 5);
    gemm(catb, attn_fs.Wm, attn_fs.bm, nullptr, mbuf, BCONST, 256, 768, 768, 256, 1);
    k_attnscore<<<BCONST, 96, 0, stream>>>(mbuf, attn_fs.Wa, attn_fs.ba, aw_fs);
    k_fuse3<<<cdiv(BCONST*DMODEL, 256), 256, 0, stream>>>(allf + 1*256, allf + 3*256, allf + 5*256,
                                                          1536, aw_fs, a_fs);

    // heads
    k_concat2<<<cdiv(BCONST*512, 256), 256, 0, stream>>>(a_cs, a_fs, cat2);
    gemm(cat2, fc_all_W, fc_all_b, nullptr, fea_cfs, BCONST, 256, 512, 512, 256, 0);
    // f_t/f_a/f_v use contiguous [b, 2i:2i+2, :] slices of allf (lda=1536)
    gemm(allf + 0,    fc_ts_W, fc_ts_b, nullptr, f_tb, BCONST, 256, 512, 1536, 256, 0);
    gemm(allf + 512,  fc_as_W, fc_as_b, nullptr, f_ab, BCONST, 256, 512, 1536, 256, 0);
    gemm(allf + 1024, fc_vs_W, fc_vs_b, nullptr, f_vb, BCONST, 256, 512, 1536, 256, 0);
    k_fuse3<<<cdiv(BCONST*DMODEL, 256), 256, 0, stream>>>(f_tb, f_ab, f_vb, 256, aw_cs, feas);

    k_logits<<<BCONST, 224, 0, stream>>>(fea_cfs, feas, cls_W, cls_b, out_logits);
}
